// RotarySelfAttention_34866544509342
// MI455X (gfx1250) — matrix operand order repaired and checked by reading, not compiled, not run on hardware
//
#include <hip/hip_runtime.h>
#include <stdint.h>

// Rotary self-attention, CDNA5 (gfx1250), bf16 WMMA (v_wmma_f32_16x16x32_bf16)
// + async global->LDS K/V staging (GLOBAL_LOAD_ASYNC_TO_LDS_B128, ASYNCcnt)
// B=2 T=2048 DIM=1024 H=16 HD=64

#define DIMC 1024
#define NH   16
#define HD   64
#define BB   2
#define TT   2048
#define BT   (BB * TT)   // 4096 rows
#define BHC  (BB * NH)   // 32 (batch*head)

typedef unsigned short u16;
typedef __attribute__((ext_vector_type(16))) __bf16 bf16x16;
typedef __attribute__((ext_vector_type(8)))  float  f32x8;

union Frag {
  bf16x16 v;
  uint4   q[2];
  u16     h[16];
};

__device__ __forceinline__ f32x8 zero8() {
  f32x8 z = {0.f, 0.f, 0.f, 0.f, 0.f, 0.f, 0.f, 0.f};
  return z;
}

__device__ __forceinline__ u16 f2bf(float x) {
  unsigned int u = __float_as_uint(x);
  unsigned int r = u + 0x7FFFu + ((u >> 16) & 1u);  // round-to-nearest-even
  return (u16)(r >> 16);
}

__device__ __forceinline__ f32x8 wmma_bf16(const Frag& a, const Frag& b, f32x8 c) {
  return __builtin_amdgcn_wmma_f32_16x16x32_bf16(false, a.v, false, b.v,
                                                 (short)0, c, false, false);
}

// async DMA: 16 bytes global -> LDS per lane, tracked by ASYNCcnt
__device__ __forceinline__ void async_ld16(uint32_t lds_off, const void* gptr) {
  asm volatile("global_load_async_to_lds_b128 %0, %1, off"
               :: "v"(lds_off), "v"((unsigned long long)(uintptr_t)gptr)
               : "memory");
}
__device__ __forceinline__ void wait_async0() {
  asm volatile("s_wait_asynccnt 0x0" ::: "memory");
}

// ---------------- conversion kernels ----------------

__global__ void k_f32_to_bf16(const float* __restrict__ src, u16* __restrict__ dst, int n) {
  int i = blockIdx.x * blockDim.x + threadIdx.x;
  if (i < n) dst[i] = f2bf(src[i]);
}

// src [K,N] f32 -> dst [N,K] bf16  (so WMMA B-operand reads are K-contiguous)
__global__ void k_transpose_bf16(const float* __restrict__ src, u16* __restrict__ dst,
                                 int K, int N) {
  int i = blockIdx.x * blockDim.x + threadIdx.x;
  if (i < K * N) {
    int k = i / N, n = i % N;
    dst[(size_t)n * K + k] = f2bf(src[i]);
  }
}

// ---------------- QKV GEMM + rotary + layout ----------------
// One wave computes a 16(M) x 64(N) tile. N-tile of 64 == one (q/k/v, head).
__global__ __launch_bounds__(128)
void k_qkv_rotary(const u16* __restrict__ xbf, const u16* __restrict__ wT,
                  const float* __restrict__ bias,
                  u16* __restrict__ qo, u16* __restrict__ ko, u16* __restrict__ vTo) {
  const int lane = threadIdx.x & 31;
  const int wave = blockIdx.x * (blockDim.x >> 5) + (threadIdx.x >> 5);
  const int NT64 = (3 * DIMC) / 64;           // 48
  const int mt   = wave / NT64;               // 0..255
  const int nt64 = wave % NT64;
  const int l15  = lane & 15;
  const int hi   = (lane & 16) ? 1 : 0;

  f32x8 acc[4] = {zero8(), zero8(), zero8(), zero8()};
  const u16* arow = xbf + (size_t)(mt * 16 + l15) * DIMC;

  #pragma unroll 1
  for (int kc = 0; kc < DIMC; kc += 32) {
    Frag a;                                    // A 16x32: lane=row, K contiguous
    const u16* ap = arow + kc + (hi ? 16 : 0);
    a.q[0] = *(const uint4*)(ap);
    a.q[1] = *(const uint4*)(ap + 8);
    #pragma unroll
    for (int nt = 0; nt < 4; ++nt) {
      int col = nt64 * 64 + nt * 16 + l15;     // B 32x16: lane=col, K contiguous
      const u16* bp = wT + (size_t)col * DIMC + kc + (hi ? 16 : 0);
      Frag b;
      b.q[0] = *(const uint4*)(bp);
      b.q[1] = *(const uint4*)(bp + 8);
      acc[nt] = wmma_bf16(a, b, acc[nt]);
    }
  }

  const int matrix = nt64 / NH;                // 0=q 1=k 2=v
  const int head   = nt64 % NH;

  float val[4][8];
  #pragma unroll
  for (int nt = 0; nt < 4; ++nt) {
    float bv = bias[nt64 * 64 + nt * 16 + l15];
    #pragma unroll
    for (int r = 0; r < 8; ++r) val[nt][r] = acc[nt][r] + bv;
  }

  #pragma unroll
  for (int nt = 0; nt < 4; ++nt) {
    int d = nt * 16 + l15;                     // 0..63 within head
    #pragma unroll
    for (int r = 0; r < 8; ++r) {
      int row = mt * 16 + r + (hi ? 8 : 0);
      int b = row / TT, t = row % TT;
      int bh = b * NH + head;
      if (matrix < 2) {
        // rotary: partner d^32 lives in acc[nt^2] of the SAME lane
        int i = d & 31;
        float ang = (float)t * __expf(-(float)i * (9.210340371976184f / 32.0f));
        float sn, cs;
        __sincosf(ang, &sn, &cs);
        float partner = val[nt ^ 2][r];
        float res = (d < 32) ? (val[nt][r] * cs - partner * sn)
                             : (val[nt][r] * cs + partner * sn);
        if (matrix == 0) {
          res *= 0.125f;                       // fold 1/sqrt(HD) into q
          qo[((size_t)bh * TT + t) * HD + d] = f2bf(res);
        } else {
          ko[((size_t)bh * TT + t) * HD + d] = f2bf(res);
        }
      } else {
        vTo[((size_t)bh * HD + d) * TT + t] = f2bf(val[nt][r]);  // V transposed
      }
    }
  }
}

// ---------------- flash attention ----------------
// One block = 4 waves = 4 query tiles (64 queries) of the SAME (b,h).
// K/V chunks (32 keys) double-buffered in LDS via async global->LDS DMA.
__global__ __launch_bounds__(128)
void k_attn(const u16* __restrict__ qg, const u16* __restrict__ kg,
            const u16* __restrict__ vT, u16* __restrict__ og) {
  __shared__ alignas(16) u16 Kbuf[2][32 * 64];   // [key][dim]   4KB each
  __shared__ alignas(16) u16 Vbuf[2][64 * 32];   // [dim][key]   4KB each
  __shared__ alignas(16) u16 Plds[4 * 512];      // per-wave P re-layout

  const int tid   = threadIdx.x;
  const int lane  = tid & 31;
  const int wslot = tid >> 5;
  const int bh  = blockIdx.x >> 5;               // 32 query groups per (b,h)
  const int qgp = blockIdx.x & 31;
  const int qt  = qgp * 4 + wslot;               // this wave's 16-query tile
  const int l15 = lane & 15;
  const int hi  = (lane & 16) ? 1 : 0;

  const u16* kpanel = kg + (size_t)bh * TT * HD; // + kb*HD : chunk is contiguous
  const u16* vpanel = vT + (size_t)bh * HD * TT; // 64 rows of TT keys

  // stage one 32-key chunk: K 4KB contiguous; V 64 rows x 64B. 4 async/thread.
  auto stage = [&](int buf, int kb) {
    uint32_t klds = (uint32_t)(uintptr_t)(const void*)&Kbuf[buf][0];
    uint32_t vlds = (uint32_t)(uintptr_t)(const void*)&Vbuf[buf][0];
    const u16* kgp2 = kpanel + (size_t)kb * HD;
    #pragma unroll
    for (int i = 0; i < 2; ++i) {
      int s = tid + i * 128;                     // 256 x 16B segments
      async_ld16(klds + s * 16, kgp2 + s * 8);
      int d = s >> 2, part = s & 3;
      async_ld16(vlds + s * 16, vpanel + (size_t)d * TT + kb + part * 8);
    }
  };

  Frag qA[2];                                    // Q as A-operand, two K=32 chunks
  const u16* qrow = qg + ((size_t)bh * TT + qt * 16 + l15) * HD;
  #pragma unroll
  for (int c = 0; c < 2; ++c) {
    const u16* p = qrow + c * 32 + (hi ? 16 : 0);
    qA[c].q[0] = *(const uint4*)(p);
    qA[c].q[1] = *(const uint4*)(p + 8);
  }

  float m[8], l[8];
  f32x8 o[4] = {zero8(), zero8(), zero8(), zero8()};
  #pragma unroll
  for (int r = 0; r < 8; ++r) { m[r] = -1e30f; l[r] = 0.f; }

  u16* pbase = Plds + wslot * 512;

  stage(0, 0);
  const int NCH = TT / 32;                       // 64 chunks
  #pragma unroll 1
  for (int c = 0; c < NCH; ++c) {
    const int cur = c & 1;
    wait_async0();                               // this chunk's DMA done (own wave)
    __syncthreads();                             // everyone's DMA visible
    if (c + 1 < NCH) stage(cur ^ 1, (c + 1) * 32);  // overlap next DMA w/ compute

    const u16* Kc = &Kbuf[cur][0];
    const u16* Vc = &Vbuf[cur][0];

    f32x8 s0 = zero8(), s1 = zero8();
    {                                            // keys 0..15 of chunk (from LDS)
      const u16* kr = Kc + l15 * HD + (hi ? 16 : 0);
      Frag b0, b1;
      b0.q[0] = *(const uint4*)(kr);      b0.q[1] = *(const uint4*)(kr + 8);
      b1.q[0] = *(const uint4*)(kr + 32); b1.q[1] = *(const uint4*)(kr + 40);
      s0 = wmma_bf16(qA[0], b0, s0);
      s0 = wmma_bf16(qA[1], b1, s0);
    }
    {                                            // keys 16..31 of chunk
      const u16* kr = Kc + (16 + l15) * HD + (hi ? 16 : 0);
      Frag b0, b1;
      b0.q[0] = *(const uint4*)(kr);      b0.q[1] = *(const uint4*)(kr + 8);
      b1.q[0] = *(const uint4*)(kr + 32); b1.q[1] = *(const uint4*)(kr + 40);
      s1 = wmma_bf16(qA[0], b0, s1);
      s1 = wmma_bf16(qA[1], b1, s1);
    }

    float corr[8];
    #pragma unroll
    for (int r = 0; r < 8; ++r) {
      float mx = fmaxf(s0[r], s1[r]);            // row-reduce over 16-lane half
      #pragma unroll
      for (int off = 1; off < 16; off <<= 1) mx = fmaxf(mx, __shfl_xor(mx, off, 32));
      float mn = fmaxf(m[r], mx);
      corr[r] = __expf(m[r] - mn);
      m[r] = mn;
      float p0 = __expf(s0[r] - mn);
      float p1 = __expf(s1[r] - mn);
      float sum = p0 + p1;
      #pragma unroll
      for (int off = 1; off < 16; off <<= 1) sum += __shfl_xor(sum, off, 32);
      l[r] = l[r] * corr[r] + sum;
      int row = r + (hi ? 8 : 0);                // C-layout row for this lane half
      pbase[row * 32 + l15]      = f2bf(p0);
      pbase[row * 32 + 16 + l15] = f2bf(p1);
    }
    #pragma unroll
    for (int nt = 0; nt < 4; ++nt)
      #pragma unroll
      for (int r = 0; r < 8; ++r) o[nt][r] *= corr[r];

    Frag pA;                                     // re-read P as A-operand (16x32)
    {
      const u16* pp = pbase + l15 * 32 + (hi ? 16 : 0);
      pA.q[0] = *(const uint4*)(pp);
      pA.q[1] = *(const uint4*)(pp + 8);
    }
    #pragma unroll
    for (int nt = 0; nt < 4; ++nt) {
      const u16* vp = Vc + (nt * 16 + l15) * 32 + (hi ? 16 : 0);
      Frag vB;
      vB.q[0] = *(const uint4*)(vp);
      vB.q[1] = *(const uint4*)(vp + 8);
      o[nt] = wmma_bf16(pA, vB, o[nt]);
    }
    __syncthreads();                             // protect buf reuse next iter
  }

  int b = bh / NH, h = bh % NH;
  #pragma unroll
  for (int nt = 0; nt < 4; ++nt) {
    int d = nt * 16 + l15;
    #pragma unroll
    for (int r = 0; r < 8; ++r) {
      int t = qt * 16 + r + (hi ? 8 : 0);
      float v = o[nt][r] / l[r];
      og[((size_t)(b * TT + t)) * DIMC + h * HD + d] = f2bf(v);
    }
  }
}

// ---------------- output projection GEMM ----------------
__global__ __launch_bounds__(128)
void k_out_gemm(const u16* __restrict__ abf, const u16* __restrict__ wT,
                const float* __restrict__ bias, float* __restrict__ out) {
  const int lane = threadIdx.x & 31;
  const int wave = blockIdx.x * 4 + (threadIdx.x >> 5);
  const int NT64 = DIMC / 64;                  // 16
  const int mt   = wave / NT64;
  const int nt64 = wave % NT64;
  const int l15  = lane & 15;
  const int hi   = (lane & 16) ? 1 : 0;

  f32x8 acc[4] = {zero8(), zero8(), zero8(), zero8()};
  const u16* arow = abf + (size_t)(mt * 16 + l15) * DIMC;

  #pragma unroll 1
  for (int kc = 0; kc < DIMC; kc += 32) {
    Frag a;
    const u16* ap = arow + kc + (hi ? 16 : 0);
    a.q[0] = *(const uint4*)(ap);
    a.q[1] = *(const uint4*)(ap + 8);
    #pragma unroll
    for (int nt = 0; nt < 4; ++nt) {
      int col = nt64 * 64 + nt * 16 + l15;
      const u16* bp = wT + (size_t)col * DIMC + kc + (hi ? 16 : 0);
      Frag b;
      b.q[0] = *(const uint4*)(bp);
      b.q[1] = *(const uint4*)(bp + 8);
      acc[nt] = wmma_bf16(a, b, acc[nt]);
    }
  }
  #pragma unroll
  for (int nt = 0; nt < 4; ++nt) {
    int col = nt64 * 64 + nt * 16 + l15;
    float bv = bias[col];
    #pragma unroll
    for (int r = 0; r < 8; ++r) {
      int row = mt * 16 + r + (hi ? 8 : 0);
      out[(size_t)row * DIMC + col] = acc[nt][r] + bv;
    }
  }
}

// ---------------- launch ----------------
extern "C" void kernel_launch(void* const* d_in, const int* in_sizes, int n_in,
                              void* d_out, int out_size, void* d_ws, size_t ws_size,
                              hipStream_t stream) {
  (void)in_sizes; (void)n_in; (void)out_size; (void)ws_size;
  const float* x    = (const float*)d_in[0];
  const float* Wqkv = (const float*)d_in[1];
  const float* bqkv = (const float*)d_in[2];
  const float* Wout = (const float*)d_in[3];
  const float* bout = (const float*)d_in[4];
  float* out = (float*)d_out;

  char* ws = (char*)d_ws;
  size_t off = 0;
  u16* xbf  = (u16*)(ws + off); off += (size_t)BT * DIMC * 2;        // 8 MiB
  u16* qbf  = (u16*)(ws + off); off += (size_t)BHC * TT * HD * 2;    // 8 MiB
  u16* kbf  = (u16*)(ws + off); off += (size_t)BHC * TT * HD * 2;    // 8 MiB
  u16* vTbf = (u16*)(ws + off); off += (size_t)BHC * HD * TT * 2;    // 8 MiB
  u16* abf  = (u16*)(ws + off); off += (size_t)BT * DIMC * 2;        // 8 MiB
  u16* wqT  = (u16*)(ws + off); off += (size_t)DIMC * 3 * DIMC * 2;  // 6 MiB
  u16* woT  = (u16*)(ws + off); off += (size_t)DIMC * DIMC * 2;      // 2 MiB

  { int n = BT * DIMC;
    k_f32_to_bf16<<<(n + 255) / 256, 256, 0, stream>>>(x, xbf, n); }
  { int n = DIMC * 3 * DIMC;
    k_transpose_bf16<<<(n + 255) / 256, 256, 0, stream>>>(Wqkv, wqT, DIMC, 3 * DIMC); }
  { int n = DIMC * DIMC;
    k_transpose_bf16<<<(n + 255) / 256, 256, 0, stream>>>(Wout, woT, DIMC, DIMC); }

  { int waves = (BT / 16) * ((3 * DIMC) / 64);          // 12288
    k_qkv_rotary<<<waves / 4, 128, 0, stream>>>(xbf, wqT, bqkv, qbf, kbf, vTbf); }
  { int blocks = BHC * (TT / 64);                       // 1024: 4 q-tiles/block
    k_attn<<<blocks, 128, 0, stream>>>(qbf, kbf, vTbf, abf); }
  { int waves = (BT / 16) * (DIMC / 64);                // 4096
    k_out_gemm<<<waves / 4, 128, 0, stream>>>(abf, woT, bout, out); }
}
